// TransConvLayer_77197742178785
// MI455X (gfx1250) — matrix operand order repaired and checked by reading, not compiled, not run on hardware
//
#include <hip/hip_runtime.h>
#include <cstdint>

// ---------------------------------------------------------------------------
// Types for CDNA5 WMMA (wave32): D(16x16 f32) = A(16x32 bf16) x B(32x16 bf16) + C
// ---------------------------------------------------------------------------
typedef __attribute__((ext_vector_type(16))) __bf16 v16bf;
typedef __attribute__((ext_vector_type(8)))  float  v8f;

union FragBF {
    v16bf    v;
    uint32_t u[8];
    uint4    u4[2];     // two 16-byte chunks: elems 0..7 and 8..15
    uint16_t h[16];
};

union Half8 {
    uint4    q;
    uint16_t h[8];
};

// f32 -> bf16, round-half-up (+0x8000 then truncate); cheap on VALU.
__device__ __forceinline__ uint16_t f2bf1(float f) {
    return (uint16_t)((__float_as_uint(f) + 0x8000u) >> 16);
}

__device__ __forceinline__ float bf2f(uint16_t u) {
    return __uint_as_float(((uint32_t)u) << 16);
}

// Pack two f32 into two bf16 (lo = a, hi = b) in 3 VALU: two adds + v_perm_b32.
// v_perm pool: s0 supplies byte indices 4..7, s1 supplies 0..3.
// dst bytes = [a.b2, a.b3, b.b2, b.b3]  -> sel 0x07060302 with (s0=b, s1=a).
__device__ __forceinline__ uint32_t pk2(float a, float b) {
    uint32_t ua = __float_as_uint(a) + 0x8000u;
    uint32_t ub = __float_as_uint(b) + 0x8000u;
    return __builtin_amdgcn_perm(ub, ua, 0x07060302u);
}

__device__ __forceinline__ uint4 pack8(const float* f) {
    uint4 o;
    o.x = pk2(f[0], f[1]); o.y = pk2(f[2], f[3]);
    o.z = pk2(f[4], f[5]); o.w = pk2(f[6], f[7]);
    return o;
}

__device__ __forceinline__ v8f wmma_bf16(const FragBF& a, const FragBF& b, v8f c) {
    return __builtin_amdgcn_wmma_f32_16x16x32_bf16(
        /*neg_a=*/false, a.v, /*neg_b=*/false, b.v,
        /*c_mod=*/(short)0, c, /*reuse_a=*/false, /*reuse_b=*/false);
}

// A-fragment (16x32) from row-major f32: lane L<16 -> M=L, elems 0..15 = K 0..15;
// lanes >=16 -> K 16..31.
__device__ __forceinline__ void load_afrags_f32(FragBF* afr, const float* __restrict__ xr, int lg) {
    #pragma unroll
    for (int kk = 0; kk < 8; ++kk) {
        const float* p = xr + kk * 32 + lg * 16;
        float t[16];
        #pragma unroll
        for (int i = 0; i < 8; ++i) { t[i] = p[i]; t[8 + i] = p[8 + i]; }
        #pragma unroll
        for (int i = 0; i < 8; ++i) afr[kk].u[i] = pk2(t[2 * i], t[2 * i + 1]);
    }
}

// B-fragment (32x16) from row-major f32: lane n = lane%16 (its row of X),
// elems 0..15 = K = kk*32 + (lane/16)*16 + e -> 16 contiguous floats.
__device__ __forceinline__ void load_bfrags_f32(FragBF* bfr, const float* __restrict__ xr, int lg) {
    #pragma unroll
    for (int kk = 0; kk < 8; ++kk) {
        const float* p = xr + kk * 32 + lg * 16;
        float t[16];
        #pragma unroll
        for (int i = 0; i < 16; ++i) t[i] = p[i];
        #pragma unroll
        for (int i = 0; i < 8; ++i) bfr[kk].u[i] = pk2(t[2 * i], t[2 * i + 1]);
    }
}

// ---------------------------------------------------------------------------
// Kernel 0: convert weights f32 -> bf16  (3 matrices of 256x256, [out][in])
// ---------------------------------------------------------------------------
__global__ void cvt_kernel(const float* __restrict__ Wq, const float* __restrict__ Wk,
                           const float* __restrict__ Wv, uint16_t* __restrict__ Wbf) {
    int i = blockIdx.x * blockDim.x + threadIdx.x;
    if (i < 65536) {
        Wbf[i]          = f2bf1(Wq[i]);
        Wbf[65536 + i]  = f2bf1(Wk[i]);
        Wbf[131072 + i] = f2bf1(Wv[i]);
    }
}

// ---------------------------------------------------------------------------
// Kernel 1: projections.
//  Q (swapped orientation: A=Wq tile, B=Xq rows -> C lane=row, VGPR=cout):
//    row-major bf16 [N][256] stored with packed b128 stores; bias folded into
//    the WMMA C initializer.
//  K,V (A=Xs rows, B=W^T): stored TRANSPOSED bf16 [256][N] (natural packed
//    store from the C layout).  Also accumulates sum(K, rows) and sum-of-
//    squares of Q and K.
//  Block = 256 threads = 8 waves; each wave owns a 16-row tile.
// ---------------------------------------------------------------------------
__global__ __launch_bounds__(256)
void proj_kernel(const float* __restrict__ Xq, const float* __restrict__ Xs,
                 const uint16_t* __restrict__ Wbf,
                 const float* __restrict__ bq, const float* __restrict__ bk,
                 const float* __restrict__ bv,
                 uint16_t* __restrict__ Q, uint16_t* __restrict__ KT,
                 uint16_t* __restrict__ VT,
                 float* __restrict__ kssum, float* __restrict__ sumsq, int N) {
    __shared__ float s_ks[256];
    __shared__ float s_sq[2];
    const int lane = threadIdx.x & 31;
    const int wave = threadIdx.x >> 5;
    const int lm = lane & 15, lg = lane >> 4;
    const int rowBase = (blockIdx.x * 8 + wave) * 16;
    const bool active = (rowBase + 16) <= N;      // wave-uniform

    s_ks[threadIdx.x] = 0.f;
    if (threadIdx.x < 2) s_sq[threadIdx.x] = 0.f;
    __syncthreads();

    if (active) {
        FragBF frg[8];
        float sqQ = 0.f, sqK = 0.f;

        // ---------------- Q projection: C^T = Wq x Xq^T ----------------
        // B-frags: Xq rows (lane = row), contiguous 64B per fragment chunk.
        load_bfrags_f32(frg, Xq + (size_t)(rowBase + lm) * 256, lg);
        #pragma unroll 1
        for (int ct = 0; ct < 16; ++ct) {
            // bias for couts ct*16 + 8*lg + 0..7  (same across lm)
            const float4 bv0 = *reinterpret_cast<const float4*>(bq + ct * 16 + 8 * lg);
            const float4 bv1 = *reinterpret_cast<const float4*>(bq + ct * 16 + 8 * lg + 4);
            v8f acc = {bv0.x, bv0.y, bv0.z, bv0.w, bv1.x, bv1.y, bv1.z, bv1.w};
            const uint16_t* wr = Wbf + (size_t)(ct * 16 + lm) * 256 + lg * 16;
            #pragma unroll
            for (int kk = 0; kk < 8; ++kk) {
                FragBF a;   // A = Wq tile: M = cout (lm), K = cin
                a.u4[0] = *reinterpret_cast<const uint4*>(wr + kk * 32);
                a.u4[1] = *reinterpret_cast<const uint4*>(wr + kk * 32 + 8);
                acc = wmma_bf16(a, frg[kk], acc);
            }
            float vals[8];
            #pragma unroll
            for (int r = 0; r < 8; ++r) { vals[r] = acc[r]; sqQ += vals[r] * vals[r]; }
            // lane's row = rowBase+lm; couts ct*16+8*lg+0..7 contiguous
            *reinterpret_cast<uint4*>(Q + (size_t)(rowBase + lm) * 256 + ct * 16 + 8 * lg) =
                pack8(vals);
        }

        // ---------------- K & V projections (A = Xs rows) ----------------
        load_afrags_f32(frg, Xs + (size_t)(rowBase + lm) * 256, lg);

        #pragma unroll 1
        for (int ct = 0; ct < 16; ++ct) {   // K -> KT (transposed store)
            const float bias = bk[ct * 16 + lm];
            v8f acc = {bias, bias, bias, bias, bias, bias, bias, bias};
            const uint16_t* wr = Wbf + 65536 + (size_t)(ct * 16 + lm) * 256 + lg * 16;
            #pragma unroll
            for (int kk = 0; kk < 8; ++kk) {
                FragBF b;
                b.u4[0] = *reinterpret_cast<const uint4*>(wr + kk * 32);
                b.u4[1] = *reinterpret_cast<const uint4*>(wr + kk * 32 + 8);
                acc = wmma_bf16(frg[kk], b, acc);
            }
            float vals[8]; float cs = 0.f;
            #pragma unroll
            for (int r = 0; r < 8; ++r) {
                vals[r] = acc[r];
                cs += vals[r];
                sqK += vals[r] * vals[r];
            }
            *reinterpret_cast<uint4*>(KT + (size_t)(ct * 16 + lm) * N + rowBase + 8 * lg) = pack8(vals);
            atomicAdd(&s_ks[ct * 16 + lm], cs);
        }

        #pragma unroll 1
        for (int ct = 0; ct < 16; ++ct) {   // V -> VT (transposed store)
            const float bias = bv[ct * 16 + lm];
            v8f acc = {bias, bias, bias, bias, bias, bias, bias, bias};
            const uint16_t* wr = Wbf + 131072 + (size_t)(ct * 16 + lm) * 256 + lg * 16;
            #pragma unroll
            for (int kk = 0; kk < 8; ++kk) {
                FragBF b;
                b.u4[0] = *reinterpret_cast<const uint4*>(wr + kk * 32);
                b.u4[1] = *reinterpret_cast<const uint4*>(wr + kk * 32 + 8);
                acc = wmma_bf16(frg[kk], b, acc);
            }
            float vals[8];
            #pragma unroll
            for (int r = 0; r < 8; ++r) vals[r] = acc[r];
            *reinterpret_cast<uint4*>(VT + (size_t)(ct * 16 + lm) * N + rowBase + 8 * lg) = pack8(vals);
        }

        atomicAdd(&s_sq[0], sqQ);
        atomicAdd(&s_sq[1], sqK);
    }
    __syncthreads();

    atomicAdd(&kssum[threadIdx.x], s_ks[threadIdx.x]);
    if (threadIdx.x == 0) {
        atomicAdd(&sumsq[0], s_sq[0]);
        atomicAdd(&sumsq[1], s_sq[1]);
    }
}

// ---------------------------------------------------------------------------
// Kernel 2: kvs[h][m][d] = sum_rows K[row][h,m] * V[row][h,d]
// Reduction GEMM over N with K,V in transposed bf16 layout; f32 atomics out.
// grid = (chunks, 8); each wave owns one of 64 (h,mtile,dtile) tiles.
// ---------------------------------------------------------------------------
__global__ __launch_bounds__(256)
void kvs_kernel(const uint16_t* __restrict__ KT, const uint16_t* __restrict__ VT,
                float* __restrict__ kvs, int N, int rowsPerChunk) {
    const int lane = threadIdx.x & 31;
    const int wave = threadIdx.x >> 5;
    const int lm = lane & 15, lg = lane >> 4;
    const int t = blockIdx.y * 8 + wave;        // 0..63
    const int h = t >> 4, mt = (t >> 2) & 3, dt = t & 3;

    const uint16_t* ar = KT + (size_t)(h * 64 + mt * 16 + lm) * N;
    const uint16_t* br = VT + (size_t)(h * 64 + dt * 16 + lm) * N;

    int r0 = blockIdx.x * rowsPerChunk;
    int r1 = r0 + rowsPerChunk; if (r1 > N) r1 = N;

    v8f acc = {0.f, 0.f, 0.f, 0.f, 0.f, 0.f, 0.f, 0.f};
    for (int r = r0; r + 32 <= r1; r += 32) {
        FragBF a, b;
        const uint16_t* ap = ar + r + lg * 16;
        a.u4[0] = *reinterpret_cast<const uint4*>(ap);
        a.u4[1] = *reinterpret_cast<const uint4*>(ap + 8);
        const uint16_t* bp = br + r + lg * 16;
        b.u4[0] = *reinterpret_cast<const uint4*>(bp);
        b.u4[1] = *reinterpret_cast<const uint4*>(bp + 8);
        __builtin_prefetch(ap + 512, 0, 1);     // global_prefetch_b8
        __builtin_prefetch(bp + 512, 0, 1);
        acc = wmma_bf16(a, b, acc);
    }
    #pragma unroll
    for (int r = 0; r < 8; ++r)
        atomicAdd(&kvs[(size_t)(h * 64 + mt * 16 + r + 8 * lg) * 64 + dt * 16 + lm], acc[r]);
}

// ---------------------------------------------------------------------------
// Kernel 3: fold the global norm scale 1/(||q||*||k||) into kvs and ks_sum,
// pack as bf16 B-matrix pcB[h][d(80)][m(64)]:
//   d<64 : kvs[h][m][d]*s (transposed)    d==64 : ks_sum[h][m]*s    d>64 : 0
// ---------------------------------------------------------------------------
__global__ void finalize_kernel(const float* __restrict__ kvs,
                                const float* __restrict__ kssum,
                                const float* __restrict__ sumsq,
                                uint16_t* __restrict__ pcB) {
    const float s = rsqrtf(fmaxf(sumsq[0], 1e-30f)) * rsqrtf(fmaxf(sumsq[1], 1e-30f));
    for (int i = blockIdx.x * blockDim.x + threadIdx.x; i < 4 * 80 * 64;
         i += blockDim.x * gridDim.x) {
        const int m = i & 63;
        const int d = (i >> 6) % 80;
        const int h = i / (80 * 64);
        float v = 0.f;
        if (d < 64)       v = kvs[(size_t)(h * 64 + m) * 64 + d] * s;
        else if (d == 64) v = kssum[h * 64 + m] * s;
        pcB[i] = f2bf1(v);
    }
}

// ---------------------------------------------------------------------------
// Kernel 4: out[n][d] = mean_h ((q_h·kvs_h)*s + N*v) / ((q_h·ks_sum_h)*s + N)
// Per 16-row wave tile: 4 heads x 2 k-steps x 5 col-tiles WMMA (col-tile 4
// holds the normalizer column), __shfl to broadcast the per-row normalizer.
// ---------------------------------------------------------------------------
__global__ __launch_bounds__(256)
void out_kernel(const uint16_t* __restrict__ Q, const uint16_t* __restrict__ VT,
                const uint16_t* __restrict__ pcB, float* __restrict__ out, int N) {
    const int lane = threadIdx.x & 31;
    const int wave = threadIdx.x >> 5;
    const int lm = lane & 15, lg = lane >> 4;
    const int rowBase = (blockIdx.x * 8 + wave) * 16;
    if (rowBase + 16 > N) return;               // wave-uniform, no syncthreads here

    const float Nf = (float)N;
    const uint16_t* qr = Q + (size_t)(rowBase + lm) * 256;
    FragBF afr[8];
    #pragma unroll
    for (int kk = 0; kk < 8; ++kk) {
        const uint16_t* p = qr + kk * 32 + lg * 16;
        afr[kk].u4[0] = *reinterpret_cast<const uint4*>(p);
        afr[kk].u4[1] = *reinterpret_cast<const uint4*>(p + 8);
    }

    float outAcc[4][8];
    #pragma unroll
    for (int j = 0; j < 4; ++j)
        #pragma unroll
        for (int r = 0; r < 8; ++r) outAcc[j][r] = 0.f;

    #pragma unroll
    for (int h = 0; h < 4; ++h) {
        v8f acc[5];
        #pragma unroll
        for (int j = 0; j < 5; ++j) acc[j] = (v8f){0.f, 0.f, 0.f, 0.f, 0.f, 0.f, 0.f, 0.f};

        #pragma unroll
        for (int ks = 0; ks < 2; ++ks) {
            #pragma unroll
            for (int j = 0; j < 5; ++j) {
                FragBF b;
                const uint16_t* bp = pcB + (size_t)(h * 80 + j * 16 + lm) * 64 + ks * 32 + lg * 16;
                b.u4[0] = *reinterpret_cast<const uint4*>(bp);
                b.u4[1] = *reinterpret_cast<const uint4*>(bp + 8);
                acc[j] = wmma_bf16(afr[h * 2 + ks], b, acc[j]);
            }
        }

        // normalizer (col 64 = col-tile 4, lane lm==0); broadcast within lane-group
        float rden[8];
        #pragma unroll
        for (int r = 0; r < 8; ++r) {
            float nrm = __shfl(acc[4][r], lane & 16, 32);
            rden[r] = 1.f / (nrm + Nf);
        }

        #pragma unroll
        for (int j = 0; j < 4; ++j) {
            Half8 vv;
            vv.q = *reinterpret_cast<const uint4*>(
                VT + (size_t)(h * 64 + j * 16 + lm) * N + rowBase + 8 * lg);
            #pragma unroll
            for (int r = 0; r < 8; ++r) {
                float v = bf2f(vv.h[r]);
                outAcc[j][r] += (acc[j][r] + Nf * v) * rden[r];
            }
        }
    }

    #pragma unroll
    for (int j = 0; j < 4; ++j)
        #pragma unroll
        for (int r = 0; r < 8; ++r)
            out[(size_t)(rowBase + r + 8 * lg) * 64 + j * 16 + lm] = outAcc[j][r] * 0.25f;
}

// ---------------------------------------------------------------------------
// Host launcher
// ---------------------------------------------------------------------------
extern "C" void kernel_launch(void* const* d_in, const int* in_sizes, int n_in,
                              void* d_out, int out_size, void* d_ws, size_t ws_size,
                              hipStream_t stream) {
    const float* Xq = (const float*)d_in[0];
    const float* Xs = (const float*)d_in[1];
    const float* Wq = (const float*)d_in[2];
    const float* bq = (const float*)d_in[3];
    const float* Wk = (const float*)d_in[4];
    const float* bk = (const float*)d_in[5];
    const float* Wv = (const float*)d_in[6];
    const float* bv = (const float*)d_in[7];
    const int N = in_sizes[0] / 256;            // [B=1, N, C=256]
    float* out = (float*)d_out;

    char* ws = (char*)d_ws;
    const size_t oW   = 0;                            // 3 * 256*256 bf16
    const size_t oQ   = 393216;                       // N*256 bf16
    const size_t oKT  = oQ  + (size_t)N * 512;        // 256 x N bf16 (transposed)
    const size_t oVT  = oKT + (size_t)N * 512;
    const size_t oKVS = oVT + (size_t)N * 512;        // 4*64*64 f32
    const size_t oKSS = oKVS + 16384 * 4;             // 256 f32
    const size_t oSQ  = oKSS + 256 * 4;               // 2 f32
    const size_t oPCB = oSQ + 256;                    // 4*80*64 bf16
    const size_t needed = oPCB + (size_t)4 * 80 * 64 * 2;
    if (ws_size < needed) return;

    uint16_t* Wbf   = (uint16_t*)(ws + oW);
    uint16_t* Qb    = (uint16_t*)(ws + oQ);
    uint16_t* KT    = (uint16_t*)(ws + oKT);
    uint16_t* VT    = (uint16_t*)(ws + oVT);
    float*    kvs   = (float*)(ws + oKVS);
    float*    kssum = (float*)(ws + oKSS);
    float*    sumsq = (float*)(ws + oSQ);
    uint16_t* pcB   = (uint16_t*)(ws + oPCB);

    // zero the accumulators (kvs + kssum + sumsq) every launch
    hipMemsetAsync(ws + oKVS, 0, (oSQ + 8) - oKVS, stream);

    cvt_kernel<<<256, 256, 0, stream>>>(Wq, Wk, Wv, Wbf);

    const int nTiles  = (N + 15) / 16;
    const int nBlocks = (nTiles + 7) / 8;
    proj_kernel<<<nBlocks, 256, 0, stream>>>(Xq, Xs, Wbf, bq, bk, bv,
                                             Qb, KT, VT, kssum, sumsq, N);

    const int kSteps = N / 32;
    const int CH = 64;                                  // more chunks -> fill WGPs
    const int rowsPerChunk = ((kSteps + CH - 1) / CH) * 32;
    const int nChunks = (N + rowsPerChunk - 1) / rowsPerChunk;
    kvs_kernel<<<dim3(nChunks, 8), 256, 0, stream>>>(KT, VT, kvs, N, rowsPerChunk);

    finalize_kernel<<<80, 256, 0, stream>>>(kvs, kssum, sumsq, pcB);

    out_kernel<<<nBlocks, 256, 0, stream>>>(Qb, VT, pcB, out, N);
}
